// GNNResBlock_32272384262682
// MI455X (gfx1250) — matrix operand order repaired and checked by reading, not compiled, not run on hardware
//
#include <hip/hip_runtime.h>

#define NN   50000
#define EE   800000
#define CODEF 128
#define HIDF  256

typedef __bf16 bf16;
typedef __attribute__((ext_vector_type(16))) __bf16 v16bf;
typedef __attribute__((ext_vector_type(8)))  __bf16 bf16x8;
typedef __attribute__((ext_vector_type(8)))  float  v8f;

union AFrag { v16bf v; bf16x8 h[2]; };

__device__ inline bf16 f2bf(float f) {
  union { float f; unsigned u; } a; a.f = f;
  unsigned r = a.u + 0x7fffu + ((a.u >> 16) & 1u);
  unsigned short hs = (unsigned short)(r >> 16);
  return __builtin_bit_cast(bf16, hs);
}

// silu via v_rcp_f32 (fast hardware reciprocal) instead of IEEE division.
__device__ inline float silu(float x) {
  return x * __builtin_amdgcn_rcpf(1.0f + __expf(-x));
}

__device__ inline float wave_sum(float v) {
#pragma unroll
  for (int off = 16; off > 0; off >>= 1) v += __shfl_xor(v, off, 32);
  return v;
}

// A fragment: 16(M) x 32(K) bf16 from LDS tile with row stride ldk.
__device__ inline v16bf load_a_frag(const bf16* At, int ldk, int lane, int kb) {
  int m = lane & 15, hi = lane >> 4;
  AFrag f;
  f.h[0] = *(const bf16x8*)(At + m * ldk + kb + hi * 16);
  f.h[1] = *(const bf16x8*)(At + m * ldk + kb + hi * 16 + 8);
  return f.v;
}

// B fragment: 32(K) x 16(N) bf16 from transposed weights Wt[n][k], row stride ldk.
__device__ inline v16bf load_b_frag(const bf16* Wt, int ldk, int n0, int lane, int kb) {
  int n = n0 + (lane & 15), hi = lane >> 4;
  return *(const v16bf*)(Wt + n * ldk + kb + hi * 16);
}

__device__ inline v8f wmma_bf16(v16bf a, v16bf b, v8f c) {
  return __builtin_amdgcn_wmma_f32_16x16x32_bf16(false, a, false, b, (short)0, c, false, false);
}

// ---------------- prep: transpose + cast weights to bf16 [n][k] ----------------
__global__ void gnn_transpose_cast(const float* __restrict__ src, bf16* __restrict__ dst,
                                   int K, int Nc) {
  int idx = blockIdx.x * blockDim.x + threadIdx.x;
  if (idx < K * Nc) {
    int k = idx / Nc, n = idx - k * Nc;
    dst[n * K + k] = f2bf(src[idx]);
  }
}

// ---------------- LayerNorm1: hn (f32 + bf16) ----------------
__global__ void __launch_bounds__(256) gnn_layernorm1(
    const float* __restrict__ h, const float* __restrict__ g, const float* __restrict__ b,
    float* __restrict__ hn, bf16* __restrict__ hnbf) {
  int wave = threadIdx.x >> 5, lane = threadIdx.x & 31;
  int node = blockIdx.x * 8 + wave;
  if (node >= NN) return;
  const float* p = h + (long long)node * CODEF + lane * 4;
  float v0 = p[0], v1 = p[1], v2 = p[2], v3 = p[3];
  float mu = wave_sum(v0 + v1 + v2 + v3) * (1.0f / 128.0f);
  float d0 = v0 - mu, d1 = v1 - mu, d2 = v2 - mu, d3 = v3 - mu;
  float var = wave_sum(d0 * d0 + d1 * d1 + d2 * d2 + d3 * d3) * (1.0f / 128.0f);
  float rs = rsqrtf(var + 1e-5f);
  int c = lane * 4;
  float o0 = d0 * rs * g[c + 0] + b[c + 0];
  float o1 = d1 * rs * g[c + 1] + b[c + 1];
  float o2 = d2 * rs * g[c + 2] + b[c + 2];
  float o3 = d3 * rs * g[c + 3] + b[c + 3];
  float* q = hn + (long long)node * CODEF + c;
  q[0] = o0; q[1] = o1; q[2] = o2; q[3] = o3;
  bf16* qb = hnbf + (long long)node * CODEF + c;
  qb[0] = f2bf(o0); qb[1] = f2bf(o1); qb[2] = f2bf(o2); qb[3] = f2bf(o3);
}

// ---------------- fused edge MLP: both GEMMs + envelope + scatter-add ----------------
// Both GEMMs' weight fragments live in 256 VGPRs per wave (loop-invariant).
// LDS holds only the A/M tiles (+16 per-edge scalars). The hn-gather for the
// next tile is software-pipelined through registers behind the two GEMMs.
__global__ void __launch_bounds__(256) gnn_edge_mlp(
    const float* __restrict__ x,
    const long long* __restrict__ erow, const long long* __restrict__ ecol,
    const bf16* __restrict__ hnbf,
    const bf16* __restrict__ W1t_g, const bf16* __restrict__ W2t_g,
    const float* __restrict__ We1_g,  // (257,256) row-major; row 256 = dist weights
    const float* __restrict__ be1, const float* __restrict__ be2,
    float* __restrict__ s, float* __restrict__ cnt) {
  extern __shared__ __align__(128) char smem[];
  bf16* At = (bf16*)smem;       // 16*256
  bf16* Mt = At + 16 * 256;     // 16*256
  float* dls = (float*)(Mt + 16 * 256);  // 16
  float* cls = dls + 16;                  // 16
  int* col16 = (int*)(cls + 16);          // 16

  int tid = threadIdx.x;
  int w = tid >> 5, lane = tid & 31;
  int n0 = w * 32;
  int nc0 = n0 + (lane & 15), nc1 = nc0 + 16;

  // Loop-invariant epilogue constants (per lane).
  float b1c0 = be1[nc0], b1c1 = be1[nc1];
  float b2c0 = be2[nc0], b2c1 = be2[nc1];
  float w1c0 = We1_g[256 * 256 + nc0], w1c1 = We1_g[256 * 256 + nc1];

  // Hoist both GEMMs' B fragments into registers (loop-invariant across tiles).
  v16bf B1[16], B2[16];
  {
    int nA = nc0 * 256 + (lane >> 4) * 16;
    int nB = nc1 * 256 + (lane >> 4) * 16;
#pragma unroll
    for (int kt = 0; kt < 8; ++kt) {
      B1[kt * 2 + 0] = *(const v16bf*)(W1t_g + nA + kt * 32);
      B1[kt * 2 + 1] = *(const v16bf*)(W1t_g + nB + kt * 32);
      B2[kt * 2 + 0] = *(const v16bf*)(W2t_g + nA + kt * 32);
      B2[kt * 2 + 1] = *(const v16bf*)(W2t_g + nB + kt * 32);
    }
  }

  // Pipelined gather setup: each thread stages 32B of hn for one (edge, half).
  int ge = tid >> 4, gp = tid & 15;
  int gko = (gp & 7) * 16;
  uint4* gdst = (uint4*)(At + ge * 256 + ((gp < 8) ? 0 : 128) + gko);
  uint4 ga0, ga1;
  {
    long long b0 = (long long)blockIdx.x * 16;
    int node = (int)((gp < 8) ? erow[b0 + ge] : ecol[b0 + ge]);
    const uint4* src = (const uint4*)(hnbf + (long long)node * CODEF + gko);
    ga0 = src[0]; ga1 = src[1];
  }

  const float kpi = 3.14159265358979323846f / 5.0f;
  const int NT = EE / 16;

  for (int t = blockIdx.x; t < NT; t += gridDim.x) {
    long long base = (long long)t * 16;
    // commit staged gather for this tile
    gdst[0] = ga0; gdst[1] = ga1;
    if (tid < 16) {
      long long nbase = base + (long long)gridDim.x * 16;
      if (nbase < EE) {
        __builtin_prefetch(&erow[nbase + tid], 0, 3);
        __builtin_prefetch(&ecol[nbase + tid], 0, 3);
      }
      int r = (int)erow[base + tid], c = (int)ecol[base + tid];
      float dx = x[r * 3 + 0] - x[c * 3 + 0];
      float dy = x[r * 3 + 1] - x[c * 3 + 1];
      float dz = x[r * 3 + 2] - x[c * 3 + 2];
      float d = sqrtf(dx * dx + dy * dy + dz * dz);
      float Cv = 0.5f * (__cosf(d * kpi) + 1.0f);
      Cv = (d <= 5.0f) ? Cv : 0.0f;
      dls[tid] = d; cls[tid] = Cv; col16[tid] = c;
      unsafeAtomicAdd(&cnt[c], 1.0f);
    }
    __syncthreads();

    // stage gather for the next tile this block will process (hidden behind GEMMs)
    {
      int tn = t + gridDim.x;
      if (tn < NT) {
        long long bn = (long long)tn * 16;
        int node = (int)((gp < 8) ? erow[bn + ge] : ecol[bn + ge]);
        const uint4* src = (const uint4*)(hnbf + (long long)node * CODEF + gko);
        ga0 = src[0]; ga1 = src[1];
      }
    }

    // GEMM1: A from LDS, B from registers
    v8f acc0 = {0, 0, 0, 0, 0, 0, 0, 0}, acc1 = {0, 0, 0, 0, 0, 0, 0, 0};
#pragma unroll
    for (int kt = 0; kt < 8; ++kt) {
      v16bf a = load_a_frag(At, 256, lane, kt * 32);
      acc0 = wmma_bf16(a, B1[kt * 2 + 0], acc0);
      acc1 = wmma_bf16(a, B1[kt * 2 + 1], acc1);
    }
    {
      int mb = (lane >> 4) * 8;
#pragma unroll
      for (int r = 0; r < 8; ++r) {
        int mm = r + mb;
        float d = dls[mm];
        float v0 = acc0[r] + b1c0 + d * w1c0;
        float v1 = acc1[r] + b1c1 + d * w1c1;
        Mt[mm * 256 + nc0] = f2bf(silu(v0));
        Mt[mm * 256 + nc1] = f2bf(silu(v1));
      }
    }
    __syncthreads();

    // GEMM2: A from LDS (Mt), B from registers, envelope, scatter-add
    v8f bcc0 = {0, 0, 0, 0, 0, 0, 0, 0}, bcc1 = {0, 0, 0, 0, 0, 0, 0, 0};
#pragma unroll
    for (int kt = 0; kt < 8; ++kt) {
      v16bf a = load_a_frag(Mt, 256, lane, kt * 32);
      bcc0 = wmma_bf16(a, B2[kt * 2 + 0], bcc0);
      bcc1 = wmma_bf16(a, B2[kt * 2 + 1], bcc1);
    }
    {
      int mb = (lane >> 4) * 8;
#pragma unroll
      for (int r = 0; r < 8; ++r) {
        int mm = r + mb;
        float Cv = cls[mm];
        int cb = col16[mm] * HIDF;
        float v0 = silu(bcc0[r] + b2c0) * Cv;
        float v1 = silu(bcc1[r] + b2c1) * Cv;
        unsafeAtomicAdd(&s[cb + nc0], v0);
        unsafeAtomicAdd(&s[cb + nc1], v1);
      }
    }
    __syncthreads();
  }
}

// ---------------- aggregate: m_aggr = s / max(cnt,1) -> bf16 ----------------
__global__ void gnn_aggregate(const float* __restrict__ s, const float* __restrict__ cnt,
                              bf16* __restrict__ agg) {
  long long i = (long long)blockIdx.x * 256 + threadIdx.x;
  if (i < (long long)NN * HIDF) {
    float c = cnt[i >> 8];
    agg[i] = f2bf(s[i] / fmaxf(c, 1.0f));
  }
}

// ---------------- node GEMM 1: hd = silu([hn, aggr] @ Wn1 + bn1) ----------------
__global__ void __launch_bounds__(256) gnn_node_hidden(
    const bf16* __restrict__ hnbf, const bf16* __restrict__ aggbf,
    const bf16* __restrict__ Wn1t_g, const float* __restrict__ bn1,
    bf16* __restrict__ hd) {
  extern __shared__ __align__(128) char smem[];
  bf16* Wt = (bf16*)smem;        // 256 x 384
  bf16* At = Wt + 256 * 384;     // 16 x 384
  float* bs = (float*)(At + 16 * 384);
  int tid = threadIdx.x;
  {
    const uint4* sp = (const uint4*)Wn1t_g; uint4* dp = (uint4*)Wt;
    for (int i = tid; i < 12288; i += 256) dp[i] = sp[i];
    if (tid < 256) bs[tid] = bn1[tid];
  }
  __syncthreads();
  int w = tid >> 5, lane = tid & 31;
  int n0 = w * 32;
  for (int t = blockIdx.x; t < NN / 16; t += gridDim.x) {
    for (int idx = tid; idx < 16 * 24; idx += 256) {
      int e = idx / 24, p = idx - e * 24;
      long long node = (long long)t * 16 + e;
      const uint4* src = (p < 8) ? (const uint4*)(hnbf + node * CODEF + p * 16)
                                 : (const uint4*)(aggbf + node * HIDF + (p - 8) * 16);
      uint4* dst = (uint4*)(At + e * 384 + p * 16);
      dst[0] = src[0]; dst[1] = src[1];
    }
    __syncthreads();
    v8f acc0 = {0, 0, 0, 0, 0, 0, 0, 0}, acc1 = {0, 0, 0, 0, 0, 0, 0, 0};
#pragma unroll
    for (int kt = 0; kt < 12; ++kt) {
      int kb = kt * 32;
      v16bf a  = load_a_frag(At, 384, lane, kb);
      v16bf b0 = load_b_frag(Wt, 384, n0, lane, kb);
      v16bf b1 = load_b_frag(Wt, 384, n0 + 16, lane, kb);
      acc0 = wmma_bf16(a, b0, acc0);
      acc1 = wmma_bf16(a, b1, acc1);
    }
    {
      int nc0 = n0 + (lane & 15), nc1 = nc0 + 16;
      int mb = (lane >> 4) * 8;
#pragma unroll
      for (int r = 0; r < 8; ++r) {
        int mm = r + mb;
        long long node = (long long)t * 16 + mm;
        hd[node * HIDF + nc0] = f2bf(silu(acc0[r] + bs[nc0]));
        hd[node * HIDF + nc1] = f2bf(silu(acc1[r] + bs[nc1]));
      }
    }
    __syncthreads();
  }
}

// ---------------- node GEMM 2 + residuals + LayerNorm2 ----------------
__global__ void __launch_bounds__(128) gnn_node_update(
    const bf16* __restrict__ hdbf, const bf16* __restrict__ Wn2t_g,
    const float* __restrict__ bn2, const float* __restrict__ hn,
    const float* __restrict__ h, const float* __restrict__ g2,
    const float* __restrict__ bt2, float* __restrict__ h1, bf16* __restrict__ h2bf) {
  extern __shared__ __align__(128) char smem[];
  bf16* Wt = (bf16*)smem;        // 128 x 256
  bf16* At = Wt + 128 * 256;     // 16 x 256
  float* H1 = (float*)(At + 16 * 256);  // 16 x 128
  float* bs = H1 + 16 * 128;
  float* gs = bs + 128;
  float* bts = gs + 128;
  int tid = threadIdx.x;
  {
    const uint4* sp = (const uint4*)Wn2t_g; uint4* dp = (uint4*)Wt;
    for (int i = tid; i < 4096; i += 128) dp[i] = sp[i];
    if (tid < 128) { bs[tid] = bn2[tid]; gs[tid] = g2[tid]; bts[tid] = bt2[tid]; }
  }
  __syncthreads();
  int w = tid >> 5, lane = tid & 31;
  int n0 = w * 32;
  for (int t = blockIdx.x; t < NN / 16; t += gridDim.x) {
    for (int idx = tid; idx < 16 * 16; idx += 128) {
      int e = idx >> 4, p = idx & 15;
      long long node = (long long)t * 16 + e;
      const uint4* src = (const uint4*)(hdbf + node * HIDF + p * 16);
      uint4* dst = (uint4*)(At + e * 256 + p * 16);
      dst[0] = src[0]; dst[1] = src[1];
    }
    __syncthreads();
    v8f acc0 = {0, 0, 0, 0, 0, 0, 0, 0}, acc1 = {0, 0, 0, 0, 0, 0, 0, 0};
#pragma unroll
    for (int kt = 0; kt < 8; ++kt) {
      int kb = kt * 32;
      v16bf a  = load_a_frag(At, 256, lane, kb);
      v16bf b0 = load_b_frag(Wt, 256, n0, lane, kb);
      v16bf b1 = load_b_frag(Wt, 256, n0 + 16, lane, kb);
      acc0 = wmma_bf16(a, b0, acc0);
      acc1 = wmma_bf16(a, b1, acc1);
    }
    {
      int nc0 = n0 + (lane & 15), nc1 = nc0 + 16;
      int mb = (lane >> 4) * 8;
#pragma unroll
      for (int r = 0; r < 8; ++r) {
        int mm = r + mb;
        long long node = (long long)t * 16 + mm;
        float v0 = acc0[r] + bs[nc0] + hn[node * CODEF + nc0] + h[node * CODEF + nc0];
        float v1 = acc1[r] + bs[nc1] + hn[node * CODEF + nc1] + h[node * CODEF + nc1];
        h1[node * CODEF + nc0] = v0; H1[mm * 128 + nc0] = v0;
        h1[node * CODEF + nc1] = v1; H1[mm * 128 + nc1] = v1;
      }
    }
    __syncthreads();
    // LayerNorm2 per node -> h2 bf16
    for (int e = w; e < 16; e += 4) {
      const float* p = &H1[e * 128 + lane * 4];
      float v0 = p[0], v1 = p[1], v2 = p[2], v3 = p[3];
      float mu = wave_sum(v0 + v1 + v2 + v3) * (1.0f / 128.0f);
      float d0 = v0 - mu, d1 = v1 - mu, d2 = v2 - mu, d3 = v3 - mu;
      float var = wave_sum(d0 * d0 + d1 * d1 + d2 * d2 + d3 * d3) * (1.0f / 128.0f);
      float rs = rsqrtf(var + 1e-5f);
      long long node = (long long)t * 16 + e;
      int c = lane * 4;
      bf16* qb = h2bf + node * CODEF + c;
      qb[0] = f2bf(d0 * rs * gs[c + 0] + bts[c + 0]);
      qb[1] = f2bf(d1 * rs * gs[c + 1] + bts[c + 1]);
      qb[2] = f2bf(d2 * rs * gs[c + 2] + bts[c + 2]);
      qb[3] = f2bf(d3 * rs * gs[c + 3] + bts[c + 3]);
    }
    __syncthreads();
  }
}

// ---------------- output MLP GEMM1: tmp = silu(h2 @ Wm1 + bm1) ----------------
__global__ void __launch_bounds__(256) gnn_mlp1(
    const bf16* __restrict__ h2bf, const bf16* __restrict__ Wm1t_g,
    const float* __restrict__ bm1, bf16* __restrict__ tmpbf) {
  extern __shared__ __align__(128) char smem[];
  bf16* Wt = (bf16*)smem;        // 256 x 128
  bf16* At = Wt + 256 * 128;     // 16 x 128
  float* bs = (float*)(At + 16 * 128);
  int tid = threadIdx.x;
  {
    const uint4* sp = (const uint4*)Wm1t_g; uint4* dp = (uint4*)Wt;
    for (int i = tid; i < 4096; i += 256) dp[i] = sp[i];
    if (tid < 256) bs[tid] = bm1[tid];
  }
  __syncthreads();
  int w = tid >> 5, lane = tid & 31;
  int n0 = w * 32;
  for (int t = blockIdx.x; t < NN / 16; t += gridDim.x) {
    for (int idx = tid; idx < 16 * 8; idx += 256) {
      int e = idx >> 3, p = idx & 7;
      long long node = (long long)t * 16 + e;
      const uint4* src = (const uint4*)(h2bf + node * CODEF + p * 16);
      uint4* dst = (uint4*)(At + e * 128 + p * 16);
      dst[0] = src[0]; dst[1] = src[1];
    }
    __syncthreads();
    v8f acc0 = {0, 0, 0, 0, 0, 0, 0, 0}, acc1 = {0, 0, 0, 0, 0, 0, 0, 0};
#pragma unroll
    for (int kt = 0; kt < 4; ++kt) {
      int kb = kt * 32;
      v16bf a  = load_a_frag(At, 128, lane, kb);
      v16bf b0 = load_b_frag(Wt, 128, n0, lane, kb);
      v16bf b1 = load_b_frag(Wt, 128, n0 + 16, lane, kb);
      acc0 = wmma_bf16(a, b0, acc0);
      acc1 = wmma_bf16(a, b1, acc1);
    }
    {
      int nc0 = n0 + (lane & 15), nc1 = nc0 + 16;
      int mb = (lane >> 4) * 8;
#pragma unroll
      for (int r = 0; r < 8; ++r) {
        int mm = r + mb;
        long long node = (long long)t * 16 + mm;
        tmpbf[node * HIDF + nc0] = f2bf(silu(acc0[r] + bs[nc0]));
        tmpbf[node * HIDF + nc1] = f2bf(silu(acc1[r] + bs[nc1]));
      }
    }
    __syncthreads();
  }
}

// ---------------- output MLP GEMM2 + residual: out = h1 + tmp @ Wm2 + bm2 ----------------
__global__ void __launch_bounds__(128) gnn_mlp2(
    const bf16* __restrict__ tmpbf, const bf16* __restrict__ Wm2t_g,
    const float* __restrict__ bm2, const float* __restrict__ h1,
    float* __restrict__ out) {
  extern __shared__ __align__(128) char smem[];
  bf16* Wt = (bf16*)smem;        // 128 x 256
  bf16* At = Wt + 128 * 256;     // 16 x 256
  float* bs = (float*)(At + 16 * 256);
  int tid = threadIdx.x;
  {
    const uint4* sp = (const uint4*)Wm2t_g; uint4* dp = (uint4*)Wt;
    for (int i = tid; i < 4096; i += 128) dp[i] = sp[i];
    if (tid < 128) bs[tid] = bm2[tid];
  }
  __syncthreads();
  int w = tid >> 5, lane = tid & 31;
  int n0 = w * 32;
  for (int t = blockIdx.x; t < NN / 16; t += gridDim.x) {
    for (int idx = tid; idx < 16 * 16; idx += 128) {
      int e = idx >> 4, p = idx & 15;
      long long node = (long long)t * 16 + e;
      const uint4* src = (const uint4*)(tmpbf + node * HIDF + p * 16);
      uint4* dst = (uint4*)(At + e * 256 + p * 16);
      dst[0] = src[0]; dst[1] = src[1];
    }
    __syncthreads();
    v8f acc0 = {0, 0, 0, 0, 0, 0, 0, 0}, acc1 = {0, 0, 0, 0, 0, 0, 0, 0};
#pragma unroll
    for (int kt = 0; kt < 8; ++kt) {
      int kb = kt * 32;
      v16bf a  = load_a_frag(At, 256, lane, kb);
      v16bf b0 = load_b_frag(Wt, 256, n0, lane, kb);
      v16bf b1 = load_b_frag(Wt, 256, n0 + 16, lane, kb);
      acc0 = wmma_bf16(a, b0, acc0);
      acc1 = wmma_bf16(a, b1, acc1);
    }
    {
      int nc0 = n0 + (lane & 15), nc1 = nc0 + 16;
      int mb = (lane >> 4) * 8;
#pragma unroll
      for (int r = 0; r < 8; ++r) {
        int mm = r + mb;
        long long node = (long long)t * 16 + mm;
        out[node * CODEF + nc0] = h1[node * CODEF + nc0] + acc0[r] + bs[nc0];
        out[node * CODEF + nc1] = h1[node * CODEF + nc1] + acc1[r] + bs[nc1];
      }
    }
    __syncthreads();
  }
}

extern "C" void kernel_launch(void* const* d_in, const int* in_sizes, int n_in,
                              void* d_out, int out_size, void* d_ws, size_t ws_size,
                              hipStream_t stream) {
  (void)in_sizes; (void)n_in; (void)out_size; (void)ws_size;
  const float* x   = (const float*)d_in[0];
  const float* h   = (const float*)d_in[1];
  const long long* ei = (const long long*)d_in[2];
  const float* We1 = (const float*)d_in[3];
  const float* be1 = (const float*)d_in[4];
  const float* We2 = (const float*)d_in[5];
  const float* be2 = (const float*)d_in[6];
  const float* Wn1 = (const float*)d_in[7];
  const float* bn1 = (const float*)d_in[8];
  const float* Wn2 = (const float*)d_in[9];
  const float* bn2 = (const float*)d_in[10];
  const float* Wm1 = (const float*)d_in[11];
  const float* bm1 = (const float*)d_in[12];
  const float* Wm2 = (const float*)d_in[13];
  const float* bm2 = (const float*)d_in[14];
  const float* g1  = (const float*)d_in[15];
  const float* bt1 = (const float*)d_in[16];
  const float* g2  = (const float*)d_in[17];
  const float* bt2 = (const float*)d_in[18];

  char* ws = (char*)d_ws;
  size_t off = 0;
  auto take = [&](size_t bytes) -> char* {
    char* p = ws + off;
    off = (off + bytes + 255) & ~(size_t)255;
    return p;
  };
  float* hn    = (float*)take((size_t)NN * CODEF * 4);
  bf16*  hnbf  = (bf16*) take((size_t)NN * CODEF * 2);
  bf16*  aggbf = (bf16*) take((size_t)NN * HIDF * 2);
  bf16*  hdbf  = (bf16*) take((size_t)NN * HIDF * 2);
  float* h1    = (float*)take((size_t)NN * CODEF * 4);
  bf16*  h2bf  = (bf16*) take((size_t)NN * CODEF * 2);
  bf16*  tmpbf = (bf16*) take((size_t)NN * HIDF * 2);
  float* sbuf  = (float*)take((size_t)NN * HIDF * 4);
  float* cnt   = (float*)take((size_t)NN * 4);
  bf16* W1t  = (bf16*)take(256 * 256 * 2);
  bf16* W2t  = (bf16*)take(256 * 256 * 2);
  bf16* Wn1t = (bf16*)take(384 * 256 * 2);
  bf16* Wn2t = (bf16*)take(256 * 128 * 2);
  bf16* Wm1t = (bf16*)take(128 * 256 * 2);
  bf16* Wm2t = (bf16*)take(256 * 128 * 2);

  hipMemsetAsync(sbuf, 0, (size_t)NN * HIDF * 4, stream);
  hipMemsetAsync(cnt, 0, (size_t)NN * 4, stream);

  gnn_transpose_cast<<<(256 * 256 + 255) / 256, 256, 0, stream>>>(We1, W1t, 256, 256);
  gnn_transpose_cast<<<(256 * 256 + 255) / 256, 256, 0, stream>>>(We2, W2t, 256, 256);
  gnn_transpose_cast<<<(384 * 256 + 255) / 256, 256, 0, stream>>>(Wn1, Wn1t, 384, 256);
  gnn_transpose_cast<<<(256 * 128 + 255) / 256, 256, 0, stream>>>(Wn2, Wn2t, 256, 128);
  gnn_transpose_cast<<<(128 * 256 + 255) / 256, 256, 0, stream>>>(Wm1, Wm1t, 128, 256);
  gnn_transpose_cast<<<(256 * 128 + 255) / 256, 256, 0, stream>>>(Wm2, Wm2t, 256, 128);

  gnn_layernorm1<<<(NN + 7) / 8, 256, 0, stream>>>(h, g1, bt1, hn, hnbf);

  size_t smem_edge = 8192 + 8192 + 64 + 64 + 64;
  gnn_edge_mlp<<<1024, 256, smem_edge, stream>>>(x, ei, ei + EE, hnbf, W1t, W2t,
                                                 We1, be1, be2, sbuf, cnt);

  gnn_aggregate<<<(int)(((size_t)NN * HIDF + 255) / 256), 256, 0, stream>>>(sbuf, cnt, aggbf);

  size_t smem_nh = 196608 + 12288 + 1024;
  gnn_node_hidden<<<512, 256, smem_nh, stream>>>(hnbf, aggbf, Wn1t, bn1, hdbf);

  size_t smem_nu = 65536 + 8192 + 8192 + 512 + 512 + 512;
  gnn_node_update<<<512, 128, smem_nu, stream>>>(hdbf, Wn2t, bn2, hn, h, g2, bt2, h1, h2bf);

  size_t smem_m1 = 65536 + 4096 + 1024;
  gnn_mlp1<<<512, 256, smem_m1, stream>>>(h2bf, Wm1t, bm1, tmpbf);

  size_t smem_m2 = 65536 + 8192 + 512;
  gnn_mlp2<<<512, 128, smem_m2, stream>>>(tmpbf, Wm2t, bm2, h1, (float*)d_out);
}
